// VQAutoEncoder_89017492177397
// MI455X (gfx1250) — matrix operand order repaired and checked by reading, not compiled, not run on hardware
//
#include <hip/hip_runtime.h>
#include <hip/hip_bf16.h>
#include <math.h>

// ---- problem constants (match reference) ----
#define D_     512
#define H_     8
#define HD_    64
#define DEPTH_ 6
#define DFF_   2048
#define G_     32
#define GD_    16
#define KC_    2048
#define Q_     2
#define B_     8
#define T_     512
#define IN_    58
#define M_     (B_*T_)   // 4096 tokens

#if defined(__has_builtin)
#if __has_builtin(__builtin_amdgcn_tensor_load_to_lds)
#define HAVE_TDM 1
#endif
#endif

typedef __attribute__((ext_vector_type(16))) __bf16 v16bf;
typedef __attribute__((ext_vector_type(4)))  __bf16 v4bf;
typedef __attribute__((ext_vector_type(8)))  float  v8f;
typedef __attribute__((ext_vector_type(4)))  unsigned int v4u;
typedef __attribute__((ext_vector_type(8)))  int    v8i;
typedef __attribute__((ext_vector_type(4)))  int    v4i;

static __device__ __forceinline__ v8f wmma_bf16(v16bf a, v16bf b, v8f c) {
    // D = A(16x32 bf16) * B(32x16 bf16) + C(16x16 f32)
    return __builtin_amdgcn_wmma_f32_16x16x32_bf16(
        false, a, false, b, (short)0, c, false, false);
}

#ifdef HAVE_TDM
// Issue a TDM 2D tile load: tile_w (f32 elems) x tile_h rows, row stride
// `stride_elems` f32, from gptr into LDS at byte offset lds_off.
// D# layout per CDNA5 ISA (group0 128b, group1 256b).
// This toolchain exposes the 6-arg builtin: (g0, g1, g2, g3, g_ext, cpol).
static __device__ __forceinline__ void tdm_load_2d(unsigned int lds_off,
                                                   const float* gptr,
                                                   int tile_w, int tile_h,
                                                   int stride_elems) {
    unsigned long long ga = (unsigned long long)(uintptr_t)gptr;
    v4u g0;
    g0[0] = 1u;                                   // count=1 (valid user D#)
    g0[1] = lds_off;                              // lds_addr (bytes)
    g0[2] = (unsigned int)ga;                     // global_addr[31:0]
    g0[3] = (unsigned int)(ga >> 32) | 0x80000000u; // addr[56:32] | type=2
    v8i g1;
    g1[0] = (2 << 16);                            // data_size = 4 bytes
    g1[1] = (int)((stride_elems & 0xffff) << 16); // tensor_dim0[15:0]
    g1[2] = (int)(((stride_elems >> 16) & 0xffff) | (4096 << 16)); // dim0 hi | dim1 lo
    g1[3] = (int)(tile_w << 16);                  // tensor_dim1 hi=0 | tile_dim0
    g1[4] = tile_h;                               // tile_dim1 | tile_dim2=0
    g1[5] = stride_elems;                         // tensor_dim0_stride[31:0]
    g1[6] = 0;                                    // stride0 hi | stride1 lo
    g1[7] = 0;
    v4i z4 = {};
    v8i z8 = {};
    __builtin_amdgcn_tensor_load_to_lds(g0, g1, z4, z4, z8, 0);
}
#endif

// =====================================================================
// Weight pre-conversion f32 -> bf16 (weights are L2-resident: convert
// once per layer, halve L2 traffic, remove cvt from GEMM inner loop).
// =====================================================================
__global__ __launch_bounds__(256)
void k_cvt_bf16(const float* __restrict__ in, __bf16* __restrict__ out) {
    const int i = blockIdx.x * 256 + threadIdx.x;
    const float4 v = reinterpret_cast<const float4*>(in)[i];
    v4bf o;
    o[0] = (__bf16)v.x; o[1] = (__bf16)v.y; o[2] = (__bf16)v.z; o[3] = (__bf16)v.w;
    reinterpret_cast<v4bf*>(out)[i] = o;
}

// =====================================================================
// GEMM: C[M,N] = act(A[M,K] @ Wb[N,K]^T + bias[N]),  Wb already bf16.
// 256 thr = 8 waves, wave tile 16x64, block tile 64(M) x 128(N).
// =====================================================================
template<int RELU>
__global__ __launch_bounds__(256)
void k_gemm(const float* __restrict__ A, const __bf16* __restrict__ W,
            const float* __restrict__ bias, float* __restrict__ C,
            int Ndim, int Kdim) {
    const int lane = threadIdx.x & 31;
    const int wid  = threadIdx.x >> 5;
    const int wm   = wid >> 1, wn = wid & 1;
    const int half = lane >> 4, l16 = lane & 15;
    const int h8   = half * 8;
    const int mrow  = blockIdx.y * 64 + wm * 16 + l16;
    const int nbase = blockIdx.x * 128 + wn * 64;

    v8f z = {};
    v8f acc[4];
#pragma unroll
    for (int t = 0; t < 4; ++t) acc[t] = z;

    for (int k0 = 0; k0 < Kdim; k0 += 32) {
        // ---- A fragment: activations f32 -> bf16 (16 cvt per 4 WMMA) ----
        const float4* ap = reinterpret_cast<const float4*>(
            A + (size_t)mrow * Kdim + k0 + half * 16);
        float af[16];
        *reinterpret_cast<float4*>(&af[0])  = ap[0];
        *reinterpret_cast<float4*>(&af[4])  = ap[1];
        *reinterpret_cast<float4*>(&af[8])  = ap[2];   // +8 floats
        *reinterpret_cast<float4*>(&af[12]) = ap[3];
        __builtin_prefetch(A + (size_t)mrow * Kdim + k0 + 64, 0, 1);
        v16bf a;
#pragma unroll
        for (int j = 0; j < 16; ++j) a[j] = (__bf16)af[j];

        // ---- 4 B fragments: direct 32B bf16 loads, zero conversion ----
#pragma unroll
        for (int t = 0; t < 4; ++t) {
            const int nrow = nbase + t * 16 + l16;
            const v16bf b = *reinterpret_cast<const v16bf*>(
                W + (size_t)nrow * Kdim + k0 + half * 16);
            acc[t] = wmma_bf16(a, b, acc[t]);
        }
    }

#pragma unroll
    for (int t = 0; t < 4; ++t) {
        const int n = nbase + t * 16 + l16;
        const float bv = bias[n];
#pragma unroll
        for (int r = 0; r < 8; ++r) {
            const int m = blockIdx.y * 64 + wm * 16 + r + h8;
            float v = acc[t][r] + bv;
            if (RELU) v = fmaxf(v, 0.0f);
            C[(size_t)m * Ndim + n] = v;
        }
    }
}

// =====================================================================
// Flash-style attention: grid = (T/128, H, B); 8 waves, wave = 16 queries.
// K/V tiles DMA'd by the Tensor Data Mover into double-buffered LDS
// staging (f32), repacked to bf16 (V transposed), WMMA for S and PV.
// =====================================================================
__global__ __launch_bounds__(256)
void k_attn(const float* __restrict__ QKV, float* __restrict__ AO) {
#ifdef HAVE_TDM
    __shared__ float  sStage[2][2][128 * 64];  // [buf][K/V][key*64+d] f32
#endif
    __shared__ __bf16 sK[128 * 64];            // [key][d]
    __shared__ __bf16 sV[64 * 128];            // [d][key]  (transposed)
    __shared__ __bf16 sP[8 * 16 * 32];         // per-wave P staging

    const int lane = threadIdx.x & 31, wid = threadIdx.x >> 5;
    const int half = lane >> 4, l16 = lane & 15, h8 = half * 8;
    const int b = blockIdx.z, h = blockIdx.y;
    const int qblk = blockIdx.x * 8 + wid;     // 0..31

    // Q fragments (two 16x32 K-steps over HD=64), softmax scale folded in
    const float* qp = QKV + ((size_t)(b * T_ + qblk * 16 + l16)) * (3 * D_) + h * HD_;
    v16bf qf[2];
#pragma unroll
    for (int s = 0; s < 2; ++s) {
#pragma unroll
        for (int j = 0; j < 8; ++j) {
            qf[s][j]     = (__bf16)(qp[s * 32 + half * 16 + j]      * 0.125f);
            qf[s][8 + j] = (__bf16)(qp[s * 32 + half * 16 + 8 + j] * 0.125f);
        }
    }

    v8f zv = {};
    v8f o[4];
#pragma unroll
    for (int t = 0; t < 4; ++t) o[t] = zv;
    float rmax[8], rsum[8];
#pragma unroll
    for (int r = 0; r < 8; ++r) { rmax[r] = -1e30f; rsum[r] = 0.0f; }

#ifdef HAVE_TDM
    // kick off DMA of key-group 0 (wave 0 issues; EXEC ignored by TDM)
    if (threadIdx.x < 32) {
        const float* kb = QKV + (size_t)(b * T_) * (3 * D_) + D_ + h * HD_;
        tdm_load_2d((unsigned int)(uintptr_t)&sStage[0][0][0], kb,       64, 128, 3 * D_);
        tdm_load_2d((unsigned int)(uintptr_t)&sStage[0][1][0], kb + D_,  64, 128, 3 * D_);
    }
#endif

    for (int kg = 0; kg < 4; ++kg) {           // 4 key groups of 128
#ifdef HAVE_TDM
        if (threadIdx.x < 32) __builtin_amdgcn_s_wait_tensorcnt(0);
        __syncthreads();
        // prefetch next group into the other buffer while we compute
        if (threadIdx.x < 32 && kg < 3) {
            const float* kb = QKV + (size_t)(b * T_ + (kg + 1) * 128) * (3 * D_)
                              + D_ + h * HD_;
            const int nb = (kg + 1) & 1;
            tdm_load_2d((unsigned int)(uintptr_t)&sStage[nb][0][0], kb,      64, 128, 3 * D_);
            tdm_load_2d((unsigned int)(uintptr_t)&sStage[nb][1][0], kb + D_, 64, 128, 3 * D_);
        }
        // repack staged f32 -> bf16 (K row-major, V transposed)
        {
            const float* stK = &sStage[kg & 1][0][0];
            const float* stV = &sStage[kg & 1][1][0];
            for (int e = threadIdx.x; e < 128 * 64; e += 256) {
                const int key = e >> 6, d = e & 63;
                sK[e]             = (__bf16)stK[e];
                sV[d * 128 + key] = (__bf16)stV[e];
            }
        }
        __syncthreads();
#else
        __syncthreads();
        for (int e = threadIdx.x; e < 128 * 64; e += 256) {
            const int key = e >> 6, d = e & 63;
            const float* base = QKV + ((size_t)(b * T_ + kg * 128 + key)) * (3 * D_)
                                + h * HD_ + d;
            sK[key * 64 + d]  = (__bf16)base[D_];
            sV[d * 128 + key] = (__bf16)base[2 * D_];
        }
        __syncthreads();
#endif

        for (int kc = 0; kc < 4; ++kc) {       // 32-key chunks
            // ---- S = Q K^T : two 16x16 tiles (keys 0..15, 16..31) ----
            v8f s0 = zv, s1 = zv;
#pragma unroll
            for (int s = 0; s < 2; ++s) {
                const int dbase = s * 32 + half * 16;
                const __bf16* k0p = &sK[(kc * 32 + l16) * 64 + dbase];
                const __bf16* k1p = &sK[(kc * 32 + 16 + l16) * 64 + dbase];
                v16bf kb0, kb1;
#pragma unroll
                for (int j = 0; j < 16; ++j) { kb0[j] = k0p[j]; kb1[j] = k1p[j]; }
                s0 = wmma_bf16(qf[s], kb0, s0);
                s1 = wmma_bf16(qf[s], kb1, s1);
            }

            // ---- online softmax: row stats via 16-lane butterflies ----
#pragma unroll
            for (int r = 0; r < 8; ++r) {
                float x0 = s0[r], x1 = s1[r];
                float mx = fmaxf(x0, x1);
#pragma unroll
                for (int off = 8; off >= 1; off >>= 1)
                    mx = fmaxf(mx, __shfl_xor(mx, off, 16));
                const float nm = fmaxf(rmax[r], mx);
                const float al = __expf(rmax[r] - nm);
                const float p0 = __expf(x0 - nm);
                const float p1 = __expf(x1 - nm);
                float ps = p0 + p1;
#pragma unroll
                for (int off = 8; off >= 1; off >>= 1)
                    ps += __shfl_xor(ps, off, 16);
                rsum[r] = rsum[r] * al + ps;
                rmax[r] = nm;
#pragma unroll
                for (int t = 0; t < 4; ++t) o[t][r] *= al;
                __bf16* pw = &sP[wid * 512 + (r + h8) * 32 + l16];
                pw[0]  = (__bf16)p0;
                pw[16] = (__bf16)p1;
            }
            asm volatile("s_wait_dscnt 0" ::: "memory");

            // ---- reload P in A-fragment layout ----
            v16bf pf;
            const __bf16* pr = &sP[wid * 512 + l16 * 32];
#pragma unroll
            for (int j = 0; j < 8; ++j) { pf[j] = pr[half * 16 + j]; pf[8 + j] = pr[half * 16 + 8 + j]; }

            // ---- O += P V  (B fragments contiguous from transposed V) ----
#pragma unroll
            for (int t = 0; t < 4; ++t) {
                const __bf16* vp = &sV[(t * 16 + l16) * 128 + kc * 32 + half * 16];
                v16bf vb;
#pragma unroll
                for (int j = 0; j < 16; ++j) vb[j] = vp[j];
                o[t] = wmma_bf16(pf, vb, o[t]);
            }
        }
    }

    // ---- normalize + store ----
#pragma unroll
    for (int t = 0; t < 4; ++t) {
#pragma unroll
        for (int r = 0; r < 8; ++r) {
            const int m = b * T_ + qblk * 16 + r + h8;
            const int c = h * HD_ + t * 16 + l16;
            AO[(size_t)m * D_ + c] = o[t][r] / rsum[r];
        }
    }
}

// =====================================================================
// Residual + LayerNorm: O[row] = LN(X[row]+Y[row]) * g + b   (in-place OK)
// =====================================================================
__global__ __launch_bounds__(256)
void k_ln(const float* __restrict__ X, const float* __restrict__ Y,
          const float* __restrict__ gam, const float* __restrict__ bet,
          float* __restrict__ O) {
    __shared__ float red[2][8];
    const int row = blockIdx.x, t = threadIdx.x;
    const int lane = t & 31, wid = t >> 5;
    const float* xr = X + (size_t)row * D_;
    const float* yr = Y + (size_t)row * D_;
    const float v0 = xr[t] + yr[t];
    const float v1 = xr[t + 256] + yr[t + 256];
    float s = v0 + v1, s2 = v0 * v0 + v1 * v1;
#pragma unroll
    for (int off = 16; off >= 1; off >>= 1) {
        s  += __shfl_xor(s,  off, 32);
        s2 += __shfl_xor(s2, off, 32);
    }
    if (lane == 0) { red[0][wid] = s; red[1][wid] = s2; }
    __syncthreads();
    if (wid == 0) {
        float a = red[0][lane & 7], b2 = red[1][lane & 7];
#pragma unroll
        for (int off = 4; off >= 1; off >>= 1) {
            a  += __shfl_xor(a,  off, 8);
            b2 += __shfl_xor(b2, off, 8);
        }
        if (lane == 0) { red[0][0] = a; red[1][0] = b2; }
    }
    __syncthreads();
    const float mean = red[0][0] * (1.0f / D_);
    const float var  = red[1][0] * (1.0f / D_) - mean * mean;
    const float rs   = rsqrtf(var + 1e-5f);
    O[(size_t)row * D_ + t]       = (v0 - mean) * rs * gam[t] + bet[t];
    O[(size_t)row * D_ + t + 256] = (v1 - mean) * rs * gam[t + 256] + bet[t + 256];
}

// =====================================================================
// Input projection + sinusoidal positional encoding (K=58, VALU is fine)
// =====================================================================
__global__ void k_inproj(const float* __restrict__ BS, const float* __restrict__ W,
                         const float* __restrict__ bias, float* __restrict__ X) {
    const int idx = blockIdx.x * blockDim.x + threadIdx.x;   // m*512 + n
    const int m = idx >> 9, n = idx & 511;
    const float* br = BS + (size_t)m * IN_;
    const float* wr = W + (size_t)n * IN_;
    float acc = bias[n];
#pragma unroll
    for (int k = 0; k < IN_; ++k) acc += br[k] * wr[k];
    const int tpos = m & (T_ - 1);
    const float div = __expf((float)(n & ~1) * (-0.017988946039015984f)); // -ln(1e4)/512
    const float ang = (float)tpos * div;
    acc += (n & 1) ? __cosf(ang) : __sinf(ang);
    X[idx] = acc;
}

// =====================================================================
// Output projection: decoded[m, n<58] = X[m] . Wn + b
// =====================================================================
__global__ void k_outproj(const float* __restrict__ X, const float* __restrict__ W,
                          const float* __restrict__ bias, float* __restrict__ OUT) {
    const int idx = blockIdx.x * blockDim.x + threadIdx.x;
    const int m = idx >> 6, n = idx & 63;
    if (n >= IN_ || m >= M_) return;
    const float* xr = X + (size_t)m * D_;
    const float* wr = W + (size_t)n * D_;
    float acc = bias[n];
    for (int k = 0; k < D_; k += 4)
        acc += xr[k] * wr[k] + xr[k + 1] * wr[k + 1]
             + xr[k + 2] * wr[k + 2] + xr[k + 3] * wr[k + 3];
    OUT[(size_t)m * IN_ + n] = acc;
}

// =====================================================================
// Grouped residual VQ: one wave per (token, group); wave-wide argmin over
// K=2048 codes; residual loop Q=2; loss accumulated into d_out via atomics.
// =====================================================================
__global__ __launch_bounds__(256)
void k_vq(const float* __restrict__ X, const float* __restrict__ CB,
          float* __restrict__ QOUT, float* __restrict__ loss) {
    const int lane = threadIdx.x & 31, wid = threadIdx.x >> 5;
    const int id = blockIdx.x * 8 + wid;          // (m,g) task
    const int m = id >> 5, g = id & 31;
    const float* xr = X + (size_t)m * D_ + g * GD_;
    float r[GD_], qsum[GD_];
#pragma unroll
    for (int j = 0; j < GD_; ++j) { r[j] = xr[j]; qsum[j] = 0.0f; }
    const float lscale = 0.1f / 2097152.0f;       // CW / (B*T*G*GD)

#pragma unroll
    for (int qi = 0; qi < Q_; ++qi) {
        const float* cb = CB + (((size_t)g * Q_ + qi) * KC_) * GD_;
        float best = 3.4e38f; int bidx = 0;
        for (int c = lane; c < KC_; c += 32) {
            const float* cp = cb + (size_t)c * GD_;
            float d = 0.0f;
#pragma unroll
            for (int j = 0; j < GD_; ++j) { const float df = r[j] - cp[j]; d += df * df; }
            if (d < best) { best = d; bidx = c; }
        }
#pragma unroll
        for (int off = 16; off >= 1; off >>= 1) {
            const float ob = __shfl_xor(best, off, 32);
            const int   oi = __shfl_xor(bidx, off, 32);
            if (ob < best || (ob == best && oi < bidx)) { best = ob; bidx = oi; }
        }
        if (lane == 0) atomicAdd(&loss[qi], best * lscale);
        const float* cq = cb + (size_t)bidx * GD_;
#pragma unroll
        for (int j = 0; j < GD_; ++j) { const float cv = cq[j]; qsum[j] += cv; r[j] -= cv; }
    }
    if (lane == 0) {
        float* qo = QOUT + (size_t)m * D_ + g * GD_;
#pragma unroll
        for (int j = 0; j < GD_; ++j) qo[j] = qsum[j];
    }
}

__global__ void k_zero(float* p, int n) {
    const int i = blockIdx.x * blockDim.x + threadIdx.x;
    if (i < n) p[i] = 0.0f;
}

// =====================================================================
extern "C" void kernel_launch(void* const* d_in, const int* in_sizes, int n_in,
                              void* d_out, int out_size, void* d_ws, size_t ws_size,
                              hipStream_t stream) {
    (void)in_sizes; (void)n_in; (void)out_size; (void)ws_size;

    const float* bs  = (const float*)d_in[0];
    const float* epw = (const float*)d_in[1];
    const float* epb = (const float*)d_in[2];
    const float* cbs = (const float*)d_in[3];
    const float* dpw = (const float*)d_in[4];
    const float* dpb = (const float*)d_in[5];
    const float* w[2][12];
    for (int p = 0; p < 2; ++p)
        for (int i = 0; i < 12; ++i)
            w[p][i] = (const float*)d_in[6 + p * 12 + i];

    float* ws = (float*)d_ws;
    const size_t MD = (size_t)M_ * D_;
    float* X   = ws;                 // M x 512
    float* Y   = ws + MD;            // M x 512
    float* AO  = ws + 2 * MD;        // M x 512
    float* QT  = ws + 3 * MD;        // M x 512 (quantized)
    float* QKV = ws + 4 * MD;        // M x 1536
    float* FF  = ws + 7 * MD;        // M x 2048
    __bf16* WB = (__bf16*)(ws + 11 * MD);  // per-layer bf16 weights (<=1M elems)

    float* dec_out = (float*)d_out;
    float* loss    = dec_out + (size_t)B_ * T_ * IN_;

    k_zero<<<1, 32, 0, stream>>>(loss, 2);
    k_inproj<<<(M_ * D_) / 256, 256, 0, stream>>>(bs, epw, epb, X);

    auto cvt = [&](const float* src, int nelem) {  // nelem multiple of 1024
        k_cvt_bf16<<<nelem / 1024, 256, 0, stream>>>(src, WB);
    };

    auto tblock = [&](float* xb, const float* const* ww) {
        for (int i = 0; i < DEPTH_; ++i) {
            const float* qkv_w = ww[0]  + (size_t)i * 3 * D_ * D_;
            const float* qkv_b = ww[1]  + (size_t)i * 3 * D_;
            const float* out_w = ww[2]  + (size_t)i * D_ * D_;
            const float* out_b = ww[3]  + (size_t)i * D_;
            const float* l1g   = ww[4]  + (size_t)i * D_;
            const float* l1b   = ww[5]  + (size_t)i * D_;
            const float* f1w   = ww[6]  + (size_t)i * DFF_ * D_;
            const float* f1b   = ww[7]  + (size_t)i * DFF_;
            const float* f2w   = ww[8]  + (size_t)i * D_ * DFF_;
            const float* f2b   = ww[9]  + (size_t)i * D_;
            const float* l2g   = ww[10] + (size_t)i * D_;
            const float* l2b   = ww[11] + (size_t)i * D_;

            cvt(qkv_w, 3 * D_ * D_);
            k_gemm<0><<<dim3(3 * D_ / 128, M_ / 64), 256, 0, stream>>>(xb, WB, qkv_b, QKV, 3 * D_, D_);
            k_attn<<<dim3(T_ / 128, H_, B_), 256, 0, stream>>>(QKV, AO);
            cvt(out_w, D_ * D_);
            k_gemm<0><<<dim3(D_ / 128, M_ / 64), 256, 0, stream>>>(AO, WB, out_b, Y, D_, D_);
            k_ln<<<M_, 256, 0, stream>>>(xb, Y, l1g, l1b, xb);
            cvt(f1w, DFF_ * D_);
            k_gemm<1><<<dim3(DFF_ / 128, M_ / 64), 256, 0, stream>>>(xb, WB, f1b, FF, DFF_, D_);
            cvt(f2w, D_ * DFF_);
            k_gemm<0><<<dim3(D_ / 128, M_ / 64), 256, 0, stream>>>(FF, WB, f2b, Y, D_, DFF_);
            k_ln<<<M_, 256, 0, stream>>>(xb, Y, l2g, l2b, xb);
        }
    };

    tblock(X, w[0]);                                     // encoder
    k_vq<<<(M_ * G_) / 8, 256, 0, stream>>>(X, cbs, QT, loss);
    tblock(QT, w[1]);                                    // decoder
    k_outproj<<<(M_ * 64) / 256, 256, 0, stream>>>(QT, dpw, dpb, dec_out);
}
